// GPT_4681514353041
// MI455X (gfx1250) — matrix operand order repaired and checked by reading, not compiled, not run on hardware
//
#include <hip/hip_runtime.h>
#include <cstddef>

typedef __bf16 bf16;
typedef __attribute__((ext_vector_type(2)))  __bf16 bf16x2;
typedef __attribute__((ext_vector_type(16))) __bf16 v16bf;
typedef __attribute__((ext_vector_type(8)))  float  v8f;
typedef __attribute__((ext_vector_type(4)))  unsigned int u32x4;

// ---------------------------------------------------------------------------
// CDNA5 WMMA fragment loaders (wave32).
// A-matrix 16x32 bf16: lane m(=lane&15) holds row m.
//   half=lane>>4: VGPR j(0..3): K = 2j + 8*half ; VGPR j(4..7): K = 8+2j + 8*half
// B-matrix 32x16 bf16 (stored in LDS as [n][k]): lane holds col n=lane&15,
//   same K order as A: VGPR j(0..3): K = 2j + 8*half ; VGPR j(4..7): K = 8+2j + 8*half
// C/D 16x16 f32: element r -> row r + 8*half, col = lane&15.
// ---------------------------------------------------------------------------
__device__ __forceinline__ v16bf load_frag_a(const bf16* base, int ld) {
  int lane = threadIdx.x & 31;
  int m = lane & 15, h = lane >> 4;
  const bf16* p = base + m * ld + 8 * h;
  v16bf a;
#pragma unroll
  for (int j = 0; j < 8; ++j) {
    int k = 2 * j + ((j >= 4) ? 8 : 0);
    bf16x2 t = *(const bf16x2*)(p + k);
    a[2 * j] = t.x;
    a[2 * j + 1] = t.y;
  }
  return a;
}

__device__ __forceinline__ v16bf load_frag_b(const bf16* base, int ld) {
  int lane = threadIdx.x & 31;
  int n = lane & 15, h = lane >> 4;
  const bf16* p = base + n * ld + 8 * h;
  v16bf b;
#pragma unroll
  for (int j = 0; j < 8; ++j) {
    bf16x2 t = *(const bf16x2*)(p + 2 * j + ((j >= 4) ? 8 : 0));
    b[2 * j] = t.x;
    b[2 * j + 1] = t.y;
  }
  return b;
}

__device__ __forceinline__ v8f wmma_bf16(v16bf a, v16bf b, v8f c) {
  return __builtin_amdgcn_wmma_f32_16x16x32_bf16(false, a, false, b, (short)0, c,
                                                 false, false);
}

// ---------------------------------------------------------------------------
// Generic 128x128-tile bf16 WMMA GEMM:  C = A[M,K] * B + bias (+gelu) (+res)
// BT=false: B is [K,N] row-major.  BT=true: B is [N,K] row-major (e.g. emb).
// K-step 64: 16 WMMAs per LDS stage. Next-tile prefetch -> global_prefetch_b8.
// ---------------------------------------------------------------------------
template <bool BT, bool HASBIAS, bool GELUF, bool RES, bool OUTF32, bool OUTBF>
__global__ __launch_bounds__(256) void gemm_wmma(
    const bf16* __restrict__ A, int lda, const bf16* __restrict__ B, int ldb,
    const float* __restrict__ bias, const float* __restrict__ res, int ldr,
    float* __restrict__ Cf, int ldc, bf16* __restrict__ Cb, int ldcb, int M,
    int N, int K) {
  __shared__ bf16 As[128 * 72];
  __shared__ bf16 Bs[128 * 72];
  const int tid = threadIdx.x;
  const int bm0 = blockIdx.y * 128, bn0 = blockIdx.x * 128;
  const int w = tid >> 5, wr = w >> 2, wc = w & 3;
  const int lane = tid & 31, nn = lane & 15, hf = lane >> 4;

  v8f acc[4][2];
  const v8f vz = {0.f, 0.f, 0.f, 0.f, 0.f, 0.f, 0.f, 0.f};
#pragma unroll
  for (int mt = 0; mt < 4; ++mt)
#pragma unroll
    for (int nt = 0; nt < 2; ++nt) acc[mt][nt] = vz;

  for (int k0 = 0; k0 < K; k0 += 64) {
    {  // A tile: 128 rows x 64 k, 32 contiguous bf16 per thread
      int r = tid >> 1, c0 = (tid & 1) * 32;
      const bf16* gsrc = A + (size_t)(bm0 + r) * lda + k0 + c0;
      const u32x4* src = (const u32x4*)gsrc;
      u32x4* dst = (u32x4*)&As[r * 72 + c0];
      dst[0] = src[0];
      dst[1] = src[1];
      dst[2] = src[2];
      dst[3] = src[3];
      if (k0 + 64 < K) __builtin_prefetch(gsrc + 64, 0, 3);
    }
    if (BT) {  // B rows are [n][k] already
      int r = tid >> 1, c0 = (tid & 1) * 32;
      int gn = bn0 + r;
      u32x4* dst = (u32x4*)&Bs[r * 72 + c0];
      if (gn < N) {
        const bf16* gsrc = B + (size_t)gn * ldb + k0 + c0;
        const u32x4* src = (const u32x4*)gsrc;
        dst[0] = src[0];
        dst[1] = src[1];
        dst[2] = src[2];
        dst[3] = src[3];
        if (k0 + 64 < K) __builtin_prefetch(gsrc + 64, 0, 3);
      } else {
        u32x4 z = 0;
        dst[0] = z;
        dst[1] = z;
        dst[2] = z;
        dst[3] = z;
      }
    } else {  // B is [K,N]: transpose into LDS [n][k]
      int kk = tid >> 2, n0 = (tid & 3) * 32;
      const bf16* src = B + (size_t)(k0 + kk) * ldb + bn0 + n0;
#pragma unroll
      for (int i = 0; i < 32; ++i) {
        bf16 v = (bn0 + n0 + i < N) ? src[i] : (bf16)0.0f;
        Bs[(n0 + i) * 72 + kk] = v;
      }
      if (k0 + 64 < K)
        __builtin_prefetch(B + (size_t)(k0 + 64 + kk) * ldb + bn0 + n0, 0, 3);
    }
    __syncthreads();

#pragma unroll
    for (int ks = 0; ks < 64; ks += 32) {
      v16bf af[4];
#pragma unroll
      for (int mt = 0; mt < 4; ++mt)
        af[mt] = load_frag_a(&As[(wr * 64 + mt * 16) * 72 + ks], 72);
      v16bf bfm[2];
#pragma unroll
      for (int nt = 0; nt < 2; ++nt)
        bfm[nt] = load_frag_b(&Bs[(wc * 32 + nt * 16) * 72 + ks], 72);
#pragma unroll
      for (int mt = 0; mt < 4; ++mt)
#pragma unroll
        for (int nt = 0; nt < 2; ++nt)
          acc[mt][nt] = wmma_bf16(af[mt], bfm[nt], acc[mt][nt]);
    }
    __syncthreads();
  }

#pragma unroll
  for (int mt = 0; mt < 4; ++mt)
#pragma unroll
    for (int nt = 0; nt < 2; ++nt) {
      int col = bn0 + wc * 32 + nt * 16 + nn;
      if (col < N) {
        float bv = HASBIAS ? bias[col] : 0.0f;
#pragma unroll
        for (int r = 0; r < 8; ++r) {
          int row = bm0 + wr * 64 + mt * 16 + r + 8 * hf;
          float val = acc[mt][nt][r] + bv;
          if (GELUF) val = 0.5f * val * (1.0f + erff(val * 0.70710678f));
          if (RES) val += res[(size_t)row * ldr + col];
          if (OUTF32) Cf[(size_t)row * ldc + col] = val;
          if (OUTBF) Cb[(size_t)row * ldcb + col] = (bf16)val;
        }
      }
    }
}

// ---------------------------------------------------------------------------
// Flash attention: one block per (q-tile of 64, head, batch). 4 waves, each
// wave owns 16 query rows. hd=64, keys iterated in tiles of 32 (causal).
// ---------------------------------------------------------------------------
__global__ __launch_bounds__(128) void attn_k(const bf16* __restrict__ qkv,
                                              bf16* __restrict__ ob) {
  const int S = 512, QKVLD = 2304;
  int qt = blockIdx.x, hh = blockIdx.y, bbx = blockIdx.z;
  int q0 = qt * 64;
  const bf16* qbase = qkv + (size_t)bbx * S * QKVLD + hh * 64;
  const bf16* kbase = qbase + 768;
  const bf16* vbase = qbase + 1536;

  __shared__ bf16 Qt[64 * 72];      // [q][hd]
  __shared__ bf16 Kt[32 * 72];      // [key][hd]  (== B layout [n][k])
  __shared__ bf16 Vt[64 * 40];      // [hd][key]  (B layout for P·V)
  __shared__ bf16 Pt[4][16 * 40];   // per-wave P in A layout [m][k]

  int tid = threadIdx.x, lane = tid & 31, w = tid >> 5;
  int nn = lane & 15, hf = lane >> 4;

  {  // load Q tile once
    int r = tid >> 1, c0 = (tid & 1) * 32;
    const u32x4* src = (const u32x4*)(qbase + (size_t)(q0 + r) * QKVLD + c0);
    u32x4* dst = (u32x4*)&Qt[r * 72 + c0];
    dst[0] = src[0];
    dst[1] = src[1];
    dst[2] = src[2];
    dst[3] = src[3];
  }

  const v8f vz = {0.f, 0.f, 0.f, 0.f, 0.f, 0.f, 0.f, 0.f};
  v8f oa[4];
#pragma unroll
  for (int i = 0; i < 4; ++i) oa[i] = vz;
  float mrow[8], lrow[8];
#pragma unroll
  for (int r = 0; r < 8; ++r) {
    mrow[r] = -1e30f;
    lrow[r] = 0.0f;
  }
  const float scale = 0.125f;  // 1/sqrt(64)
  const float NINF = -10000.0f;
  const int kend = q0 + 64;

  for (int k0 = 0; k0 < kend; k0 += 32) {
    {  // load K tile (natural) and V tile (transposed)
      int kr = tid >> 2, c0 = (tid & 3) * 16;
      const bf16* ksrc = kbase + (size_t)(k0 + kr) * QKVLD + c0;
      const u32x4* src = (const u32x4*)ksrc;
      u32x4* dst = (u32x4*)&Kt[kr * 72 + c0];
      dst[0] = src[0];
      dst[1] = src[1];
      const bf16* vs = vbase + (size_t)(k0 + kr) * QKVLD + c0;
#pragma unroll
      for (int i = 0; i < 16; ++i) Vt[(c0 + i) * 40 + kr] = vs[i];
      if (k0 + 32 < kend) {  // prefetch next K/V tile -> global_prefetch_b8
        __builtin_prefetch(ksrc + 32 * QKVLD, 0, 3);
        __builtin_prefetch(vs + 32 * QKVLD, 0, 3);
      }
    }
    __syncthreads();

    // scores: S = Q(16x64) * K^T(64x32), accumulate over hd in 2 steps
    v8f sa0 = vz, sa1 = vz;
#pragma unroll
    for (int ks = 0; ks < 64; ks += 32) {
      v16bf aq = load_frag_a(&Qt[w * 16 * 72 + ks], 72);
      v16bf bk0 = load_frag_b(&Kt[ks], 72);
      v16bf bk1 = load_frag_b(&Kt[16 * 72 + ks], 72);
      sa0 = wmma_bf16(aq, bk0, sa0);
      sa1 = wmma_bf16(aq, bk1, sa1);
    }

    // online softmax, row-wise across 16 lanes of one half
#pragma unroll
    for (int r = 0; r < 8; ++r) {
      int qrow = q0 + w * 16 + r + 8 * hf;
      float s0 = sa0[r] * scale + ((k0 + nn > qrow) ? NINF : 0.0f);
      float s1 = sa1[r] * scale + ((k0 + 16 + nn > qrow) ? NINF : 0.0f);
      float mx = fmaxf(s0, s1);
      mx = fmaxf(mx, __shfl_xor(mx, 1));
      mx = fmaxf(mx, __shfl_xor(mx, 2));
      mx = fmaxf(mx, __shfl_xor(mx, 4));
      mx = fmaxf(mx, __shfl_xor(mx, 8));
      float nm = fmaxf(mrow[r], mx);
      float al = __expf(mrow[r] - nm);
      float p0 = __expf(s0 - nm), p1 = __expf(s1 - nm);
      float rs = p0 + p1;
      rs += __shfl_xor(rs, 1);
      rs += __shfl_xor(rs, 2);
      rs += __shfl_xor(rs, 4);
      rs += __shfl_xor(rs, 8);
      lrow[r] = lrow[r] * al + rs;
      mrow[r] = nm;
#pragma unroll
      for (int n2 = 0; n2 < 4; ++n2) oa[n2][r] = oa[n2][r] * al;
      Pt[w][(r + 8 * hf) * 40 + nn] = (bf16)p0;
      Pt[w][(r + 8 * hf) * 40 + 16 + nn] = (bf16)p1;
    }
    __syncthreads();

    // O += P(16x32) * V(32x64)
    v16bf ap = load_frag_a(&Pt[w][0], 40);
#pragma unroll
    for (int n2 = 0; n2 < 4; ++n2) {
      v16bf bv = load_frag_b(&Vt[n2 * 16 * 40], 40);
      oa[n2] = wmma_bf16(ap, bv, oa[n2]);
    }
    __syncthreads();
  }

#pragma unroll
  for (int n2 = 0; n2 < 4; ++n2)
#pragma unroll
    for (int r = 0; r < 8; ++r) {
      int qrow = q0 + w * 16 + r + 8 * hf;
      int col = hh * 64 + n2 * 16 + nn;
      ob[(size_t)(bbx * S + qrow) * 768 + col] = (bf16)(oa[n2][r] / lrow[r]);
    }
}

// ---------------------------------------------------------------------------
// Elementwise / reduction helper kernels
// ---------------------------------------------------------------------------
__global__ void embed_k(const int* __restrict__ x, const float* __restrict__ emb,
                        const float* __restrict__ pos, float* __restrict__ h) {
  int row = blockIdx.x;  // b*S + s
  int s = row & 511;
  int tok = x[row];
  for (int d = threadIdx.x; d < 768; d += 256)
    h[(size_t)row * 768 + d] = emb[(size_t)tok * 768 + d] + pos[(size_t)s * 768 + d];
}

__global__ __launch_bounds__(256) void layernorm_k(
    const float* __restrict__ x, const float* __restrict__ g,
    const float* __restrict__ bb, float* __restrict__ outf,
    bf16* __restrict__ outb, int D) {
  __shared__ float red[256];
  int row = blockIdx.x, tid = threadIdx.x;
  const float* xr = x + (size_t)row * D;
  float s = 0.0f;
  for (int i = tid; i < D; i += 256) s += xr[i];
  red[tid] = s;
  __syncthreads();
  for (int o = 128; o > 0; o >>= 1) {
    if (tid < o) red[tid] += red[tid + o];
    __syncthreads();
  }
  float mean = red[0] / D;
  __syncthreads();
  float v = 0.0f;
  for (int i = tid; i < D; i += 256) {
    float d = xr[i] - mean;
    v += d * d;
  }
  red[tid] = v;
  __syncthreads();
  for (int o = 128; o > 0; o >>= 1) {
    if (tid < o) red[tid] += red[tid + o];
    __syncthreads();
  }
  float inv = rsqrtf(red[0] / D + 1e-6f);
  for (int i = tid; i < D; i += 256) {
    float y = (xr[i] - mean) * inv * g[i] + bb[i];
    if (outf) outf[(size_t)row * D + i] = y;
    outb[(size_t)row * D + i] = (bf16)y;
  }
}

__global__ __launch_bounds__(256) void softmax_k(float* __restrict__ p, int N) {
  __shared__ float red[256];
  int row = blockIdx.x, tid = threadIdx.x;
  float* r = p + (size_t)row * N;
  float mx = -3.4e38f;
  for (int i = tid; i < N; i += 256) mx = fmaxf(mx, r[i]);
  red[tid] = mx;
  __syncthreads();
  for (int o = 128; o > 0; o >>= 1) {
    if (tid < o) red[tid] = fmaxf(red[tid], red[tid + o]);
    __syncthreads();
  }
  mx = red[0];
  __syncthreads();
  float s = 0.0f;
  for (int i = tid; i < N; i += 256) {
    float e = __expf(r[i] - mx);
    r[i] = e;
    s += e;
  }
  red[tid] = s;
  __syncthreads();
  for (int o = 128; o > 0; o >>= 1) {
    if (tid < o) red[tid] += red[tid + o];
    __syncthreads();
  }
  float inv = 1.0f / red[0];
  for (int i = tid; i < N; i += 256) r[i] *= inv;
}

__global__ void cvt_k(const float* __restrict__ in, bf16* __restrict__ out,
                      size_t n) {
  size_t i = (size_t)blockIdx.x * 256 + threadIdx.x;
  size_t stride = (size_t)gridDim.x * 256;
  for (; i < n; i += stride) out[i] = (bf16)in[i];
}

// ---------------------------------------------------------------------------
extern "C" void kernel_launch(void* const* d_in, const int* in_sizes, int n_in,
                              void* d_out, int out_size, void* d_ws,
                              size_t ws_size, hipStream_t stream) {
  (void)in_sizes;
  (void)n_in;
  (void)out_size;
  (void)ws_size;
  const int B = 4, S = 512, D = 768, L = 12, V = 50257, F = 3072;
  const int BS = B * S;

  const int* x = (const int*)d_in[0];
  const float* emb = (const float*)d_in[1];
  const float* pos = (const float*)d_in[2];
  const float* ln1g = (const float*)d_in[3];
  const float* ln1b = (const float*)d_in[4];
  const float* wqkv = (const float*)d_in[5];
  const float* bqkv = (const float*)d_in[6];
  const float* wo = (const float*)d_in[7];
  const float* bo = (const float*)d_in[8];
  const float* ln2g = (const float*)d_in[9];
  const float* ln2b = (const float*)d_in[10];
  const float* wfc = (const float*)d_in[11];
  const float* bfc = (const float*)d_in[12];
  const float* wproj = (const float*)d_in[13];
  const float* bproj = (const float*)d_in[14];
  const float* lnfg = (const float*)d_in[15];
  const float* lnfb = (const float*)d_in[16];

  float* probs = (float*)d_out;               // [BS, V]
  float* hidden = probs + (size_t)BS * V;     // [BS, D]

  char* ws = (char*)d_ws;
  size_t off = 0;
  auto alloc = [&](size_t bytes) -> void* {
    void* p = ws + off;
    off += (bytes + 255) & ~(size_t)255;
    return p;
  };
  float* h = (float*)alloc((size_t)BS * D * 4);
  bf16* lnbuf = (bf16*)alloc((size_t)BS * D * 2);
  bf16* qkvb = (bf16*)alloc((size_t)BS * 3 * D * 2);
  bf16* ob = (bf16*)alloc((size_t)BS * D * 2);
  bf16* fcb = (bf16*)alloc((size_t)BS * F * 2);
  bf16* wb = (bf16*)alloc((size_t)D * F * 2);
  bf16* embb = (bf16*)alloc((size_t)V * D * 2);

  embed_k<<<BS, 256, 0, stream>>>(x, emb, pos, h);

  for (int l = 0; l < L; ++l) {
    // --- attention block ---
    layernorm_k<<<BS, 256, 0, stream>>>(h, ln1g + l * D, ln1b + l * D,
                                        (float*)nullptr, lnbuf, D);
    cvt_k<<<1024, 256, 0, stream>>>(wqkv + (size_t)l * D * 3 * D, wb,
                                    (size_t)D * 3 * D);
    gemm_wmma<false, true, false, false, false, true>
        <<<dim3(18, 16), 256, 0, stream>>>(lnbuf, D, wb, 3 * D, bqkv + l * 3 * D,
                                           (const float*)nullptr, 0,
                                           (float*)nullptr, 0, qkvb, 3 * D, BS,
                                           3 * D, D);
    attn_k<<<dim3(8, 12, 4), 128, 0, stream>>>(qkvb, ob);
    cvt_k<<<1024, 256, 0, stream>>>(wo + (size_t)l * D * D, wb, (size_t)D * D);
    gemm_wmma<false, true, false, true, true, false>
        <<<dim3(6, 16), 256, 0, stream>>>(ob, D, wb, D, bo + l * D, h, D, h, D,
                                          (bf16*)nullptr, 0, BS, D, D);
    // --- MLP block ---
    layernorm_k<<<BS, 256, 0, stream>>>(h, ln2g + l * D, ln2b + l * D,
                                        (float*)nullptr, lnbuf, D);
    cvt_k<<<1024, 256, 0, stream>>>(wfc + (size_t)l * D * F, wb, (size_t)D * F);
    gemm_wmma<false, true, true, false, false, true>
        <<<dim3(24, 16), 256, 0, stream>>>(lnbuf, D, wb, F, bfc + l * F,
                                           (const float*)nullptr, 0,
                                           (float*)nullptr, 0, fcb, F, BS, F, D);
    cvt_k<<<1024, 256, 0, stream>>>(wproj + (size_t)l * F * D, wb,
                                    (size_t)F * D);
    gemm_wmma<false, true, false, true, true, false>
        <<<dim3(6, 16), 256, 0, stream>>>(fcb, F, wb, D, bproj + l * D, h, D, h,
                                          D, (bf16*)nullptr, 0, BS, D, F);
  }

  // final layernorm -> hidden (f32 out) + bf16 copy for logits GEMM
  layernorm_k<<<BS, 256, 0, stream>>>(h, lnfg, lnfb, hidden, lnbuf, D);
  // logits = hidden @ emb^T  (emb is [V,D] == [N,K] -> BT=true)
  cvt_k<<<4096, 256, 0, stream>>>(emb, embb, (size_t)V * D);
  gemm_wmma<true, false, false, false, true, false>
      <<<dim3((V + 127) / 128, 16), 256, 0, stream>>>(
          lnbuf, D, embb, D, (const float*)nullptr, (const float*)nullptr, 0,
          probs, V, (bf16*)nullptr, 0, BS, V, D);
  softmax_k<<<BS, 256, 0, stream>>>(probs, V);
}
